// DGCNNCls_36086315221256
// MI455X (gfx1250) — matrix operand order repaired and checked by reading, not compiled, not run on hardware
//
#include <hip/hip_runtime.h>

// ---------------------------------------------------------------------------
// DGCNN classifier for MI455X (gfx1250, wave32, WMMA). Compile-time shapes.
//   - pairwise-distance Gram tiles:  V_WMMA_F32_16X16X4_F32   (f32 for top-k)
//   - edge-conv / w5 GEMMs:          V_WMMA_F32_16X16X32_F16  (f16 in, f32 acc)
// Operands for the f16 path are pre-packed (f16, fragment-permuted) so every
// A/B fragment is a single contiguous 32B load (2x b128 / 2x ds_load_b128).
// ---------------------------------------------------------------------------

typedef __attribute__((ext_vector_type(2)))  float    v2f;
typedef __attribute__((ext_vector_type(8)))  float    v8f;
typedef __attribute__((ext_vector_type(16))) _Float16 v16h;

#define K_NN    20
#define SLOPE   0.2f
#define NEG_BIG (-3.402823e38f)

// A-fragment permutation for V_WMMA_*_16x16x32_f16:
// element e of lane-half h holds K = 2*(j&3) + 16*(j>>2) + 8*h + (e&1), j=e>>1.
// apos(c) maps channel c to its packed position; block-local:
// apos(cb + r) == cb + apos(r) for 32-aligned cb, r in [0,32).
__device__ __forceinline__ int apos(int c) {
    int K    = c & 31;
    int half = (K >> 3) & 1;
    int j    = ((K >> 1) & 3) | (((K >> 4) & 1) << 2);
    int e    = 2 * j + (K & 1);
    return (c & ~31) + half * 16 + e;
}

__device__ __forceinline__ float leaky(float x) { return (x >= 0.f) ? x : SLOPE * x; }

// ---------------------------------------------------------------------------
// 0) transpose x (B,3,N) -> xt (B,N,4), channel 3 zero-padded
// ---------------------------------------------------------------------------
__global__ void transpose4_kernel(const float* __restrict__ x, float* __restrict__ xt,
                                  int B, int N) {
    int i = blockIdx.x * blockDim.x + threadIdx.x;   // over B*N
    if (i >= B * N) return;
    int b = i / N, n = i % N;
    const float* xb = x + (size_t)b * 3 * N + n;
    float4 v = make_float4(xb[0], xb[(size_t)N], xb[(size_t)2 * N], 0.f);
    *(float4*)(xt + (size_t)i * 4) = v;
}

// ---------------------------------------------------------------------------
// 1) weight repack: w (Kreal x O) f32 -> wT (O x KP) f16, K zero-padded.
//    layer1 mode: rows {0,1,2,Z,3,4,5,Z,...} to match the channel-4-padded
//    [diff(4), ctr(4)] feature layout.
// ---------------------------------------------------------------------------
__global__ void convertWT_kernel(const float* __restrict__ w, _Float16* __restrict__ wT,
                                 int Kreal, int KP, int O, int layer1) {
    int t = blockIdx.x * blockDim.x + threadIdx.x;
    if (t >= O * KP) return;
    int o = t / KP, c = t % KP;
    float v = 0.f;
    if (layer1) {
        if (c < 3)                 v = w[(size_t)c * O + o];
        else if (c >= 4 && c < 7)  v = w[(size_t)(c - 1) * O + o];
    } else if (c < Kreal) {
        v = w[(size_t)c * O + o];
    }
    wT[(size_t)o * KP + apos(c)] = (_Float16)v;
}

// ---------------------------------------------------------------------------
// 2) per-point squared norms
// ---------------------------------------------------------------------------
template <int C, int STRIDE>
__global__ void sqnorm_kernel(const float* __restrict__ xin, float* __restrict__ sqn,
                              int total) {
    int i = blockIdx.x * blockDim.x + threadIdx.x;
    if (i >= total) return;
    const float* p = xin + (size_t)i * STRIDE;
    float s = 0.f;
#pragma unroll 4
    for (int c = 0; c < C; ++c) { float v = p[c]; s += v * v; }
    sqn[i] = s;
}

// ---------------------------------------------------------------------------
// 3) kNN top-20 via f32 WMMA Gram tiles. One wave per 16-row slice.
//    pd = 2*x.x^T - |xi|^2 - |xj|^2 ; lanes 0-15 keep sorted top-20 in regs.
// ---------------------------------------------------------------------------
template <int C, int STRIDE>
__global__ void knn_kernel(const float* __restrict__ xin, const float* __restrict__ sqn,
                           int* __restrict__ idxout, int N) {
    const int  lane = threadIdx.x;
    const bool hi   = lane >= 16;
    const int  ln   = lane & 15;
    const int  tilesPerB = N >> 4;
    const int  bId  = blockIdx.x / tilesPerB;
    const int  row0 = (blockIdx.x % tilesPerB) * 16;

    const float* xb  = xin + (size_t)bId * N * STRIDE;
    const float* sqb = sqn + (size_t)bId * N;

    __shared__ __align__(16) float s_feat[16 * C];
    __shared__ float s_rowsq[16];
    __shared__ float s_tile[16][17];

    for (int t = lane; t < 16 * C; t += 32) {
        int r = t / C, c = t % C;
        s_feat[r * C + c] = xb[(size_t)(row0 + r) * STRIDE + c];
    }
    if (lane < 16) s_rowsq[lane] = sqb[row0 + lane];
    __syncthreads();

    float vals[K_NN];
    int   inds[K_NN];
#pragma unroll
    for (int i = 0; i < K_NN; ++i) { vals[i] = NEG_BIG; inds[i] = 0; }

    constexpr int STEPS = C / 4;           // C is a multiple of 4 (layer1 padded)

    for (int mt = 0; mt < tilesPerB; ++mt) {
        const int col = mt * 16 + ln;
        const float* colp = xb + (size_t)col * STRIDE;
        v8f acc = {};
#pragma unroll
        for (int s = 0; s < STEPS; ++s) {
            const int k0 = s * 4 + (hi ? 2 : 0);
            v2f a   = *(const v2f*)(s_feat + ln * C + k0);
            v2f bfr = *(const v2f*)(colp + k0);
            acc = __builtin_amdgcn_wmma_f32_16x16x4_f32(false, a, false, bfr,
                                                        (short)0, acc, false, false);
        }
        const float csq = sqb[col];
        __syncthreads();
#pragma unroll
        for (int v = 0; v < 8; ++v) {
            int r = v + (hi ? 8 : 0);
            s_tile[r][ln] = 2.f * acc[v] - s_rowsq[r] - csq;
        }
        __syncthreads();
        if (lane < 16) {
            // single copy of the insertion ladder (do NOT unroll this loop)
#pragma unroll 1
            for (int j = 0; j < 16; ++j) {
                float v = s_tile[lane][j];
                if (v > vals[K_NN - 1]) {
                    float cv = v; int ci = mt * 16 + j;
                    // branchless rotate-insert: 1 cmp + 4 cndmask per stage
#pragma unroll
                    for (int p = 0; p < K_NN; ++p) {
                        bool  sw = cv > vals[p];
                        float nv = sw ? cv : vals[p];
                        int   ni = sw ? ci : inds[p];
                        float tv = sw ? vals[p] : cv;
                        int   ti = sw ? inds[p] : ci;
                        vals[p] = nv; inds[p] = ni;
                        cv = tv; ci = ti;
                    }
                }
            }
        }
    }

    if (lane < 16) {
        const int n = row0 + lane;
#pragma unroll
        for (int j = 0; j < K_NN; ++j)
            idxout[((size_t)bId * N + n) * K_NN + j] = inds[j];
    }
}

// ---------------------------------------------------------------------------
// 4) edge-conv: one wave per point.
//    A (32 x TWOC, rows 20..31 duplicate row 0 - max-neutral) is built once in
//    LDS in fragment-permuted f16; B fragments are contiguous v16h from wT.
//    g>0 so affine+leaky applied after the k-max.
//    Outputs: f32 slice of xcat (stride CAT) and f16 fragment-permuted xcatp.
// ---------------------------------------------------------------------------
template <int STRIDE, int CPAD, int TWOC, int O>
__global__ void edge_kernel(const float* __restrict__ xin, const int* __restrict__ idx,
                            const _Float16* __restrict__ wT,
                            const float* __restrict__ g, const float* __restrict__ bb,
                            float* __restrict__ outF, _Float16* __restrict__ outP,
                            int catOff, int CAT, int N) {
    const int  lane = threadIdx.x;
    const bool hi   = lane >= 16;
    const int  ln   = lane & 15;
    const int  pid  = blockIdx.x;
    const int  bId  = pid / N;
    const int  n    = pid % N;
    const float* xb = xin + (size_t)bId * N * STRIDE;

    __shared__ __align__(32) _Float16 s_a[32 * TWOC];
    __shared__ float s_ctr[CPAD];
    __shared__ int   s_idx[K_NN];

    if (lane < K_NN) s_idx[lane] = idx[(size_t)pid * K_NN + lane];
    for (int c = lane; c < CPAD; c += 32) s_ctr[c] = xb[(size_t)n * STRIDE + c];
    __syncthreads();

    // fill A rows; permutation is block-local so hoist apos(lane)
    const int pl = apos(lane);             // position of channel (cb + lane) is cb + pl
#pragma unroll 1
    for (int k = 0; k < 32; ++k) {
        const int kk = (k < K_NN) ? k : 0;
        const float* nrow = xb + (size_t)s_idx[kk] * STRIDE;
        _Float16* arow = s_a + k * TWOC + pl;
#pragma unroll
        for (int cb = 0; cb < TWOC; cb += 32) {
            int c = cb + lane;
            float v;
            if (c < CPAD)            v = nrow[c] - s_ctr[c];
            else if (c < 2 * CPAD)   v = s_ctr[c - CPAD];
            else                     v = 0.f;
            arow[cb] = (_Float16)v;
        }
    }
    __syncthreads();

    constexpr int KSTEPS = TWOC / 32;
    constexpr int NTILES = O / 16;
    const int hoff = hi ? 16 : 0;

    for (int t = 0; t < NTILES; ++t) {
        const int o = t * 16 + ln;
        const _Float16* wrow = wT + (size_t)o * TWOC;
        v8f acc0 = {};
        v8f acc1 = {};
#pragma unroll
        for (int s = 0; s < KSTEPS; ++s) {
            v16h a0 = *(const v16h*)(s_a + ln * TWOC + s * 32 + hoff);
            v16h a1 = *(const v16h*)(s_a + (16 + ln) * TWOC + s * 32 + hoff);
            v16h bf = *(const v16h*)(wrow + s * 32 + hoff);
            acc0 = __builtin_amdgcn_wmma_f32_16x16x32_f16(false, a0, false, bf,
                                                          (short)0, acc0, false, false);
            acc1 = __builtin_amdgcn_wmma_f32_16x16x32_f16(false, a1, false, bf,
                                                          (short)0, acc1, false, false);
        }
        float m = NEG_BIG;
#pragma unroll
        for (int v = 0; v < 8; ++v) m = fmaxf(m, fmaxf(acc0[v], acc1[v]));
        m = fmaxf(m, __shfl_xor(m, 16, 32));
        if (lane < 16) {
            float h = leaky(m * g[o] + bb[o]);
            outF[(size_t)pid * CAT + o]                 = h;
            outP[(size_t)pid * CAT + apos(catOff + o)]  = (_Float16)h;
        }
    }
}

// ---------------------------------------------------------------------------
// 5) w5 GEMM: (B*N, 512) @ (512, 1024), fused affine+leaky.
//    A from fragment-permuted xcatp, B from w5T. One wave per 16x16 tile.
// ---------------------------------------------------------------------------
template <int KT, int O>
__global__ void gemm5_kernel(const _Float16* __restrict__ xcatp,
                             const _Float16* __restrict__ w5T,
                             const float* __restrict__ g5, const float* __restrict__ b5,
                             float* __restrict__ h5) {
    const int  lane = threadIdx.x;
    const bool hi   = lane >= 16;
    const int  ln   = lane & 15;
    constexpr int TILES_O = O / 16;
    const int  tM = blockIdx.x / TILES_O;
    const int  tO = blockIdx.x % TILES_O;
    const int  row = tM * 16 + ln;
    const int  o   = tO * 16 + ln;
    const int  hoff = hi ? 16 : 0;

    const _Float16* arow = xcatp + (size_t)row * KT;
    const _Float16* brow = w5T   + (size_t)o   * KT;

    v8f acc = {};
#pragma unroll
    for (int s = 0; s < KT / 32; ++s) {
        v16h a  = *(const v16h*)(arow + s * 32 + hoff);
        v16h bf = *(const v16h*)(brow + s * 32 + hoff);
        acc = __builtin_amdgcn_wmma_f32_16x16x32_f16(false, a, false, bf,
                                                     (short)0, acc, false, false);
    }
#pragma unroll
    for (int v = 0; v < 8; ++v) {
        int r = tM * 16 + v + (hi ? 8 : 0);
        h5[(size_t)r * O + o] = leaky(acc[v] * g5[o] + b5[o]);
    }
}

// ---------------------------------------------------------------------------
// 6) global pooling: (B,N,E) -> (B,2E) = [max_n, mean_n]
// ---------------------------------------------------------------------------
__global__ void pool_kernel(const float* __restrict__ h5, float* __restrict__ pooled,
                            int B, int N, int E) {
    int i = blockIdx.x * blockDim.x + threadIdx.x;
    if (i >= B * E) return;
    int b = i / E, c = i % E;
    const float* p = h5 + (size_t)b * N * E + c;
    float mx = NEG_BIG, sm = 0.f;
    for (int n = 0; n < N; ++n) {
        float v = p[(size_t)n * E];
        mx = fmaxf(mx, v);
        sm += v;
    }
    pooled[(size_t)b * 2 * E + c]     = mx;
    pooled[(size_t)b * 2 * E + E + c] = sm / (float)N;
}

// ---------------------------------------------------------------------------
// 7) small FC: out = [leaky]((in @ w [+ bias]) [* g + ab])
// ---------------------------------------------------------------------------
__global__ void fc_kernel(const float* __restrict__ in, const float* __restrict__ w,
                          const float* __restrict__ bias, const float* __restrict__ g,
                          const float* __restrict__ ab, float* __restrict__ out,
                          int B, int I, int O, int doLeaky) {
    int t = blockIdx.x * blockDim.x + threadIdx.x;
    if (t >= B * O) return;
    int b = t / O, o = t % O;
    const float* row = in + (size_t)b * I;
    float s = 0.f;
    for (int c = 0; c < I; ++c) s += row[c] * w[(size_t)c * O + o];
    if (bias) s += bias[o];
    if (g)    s = s * g[o] + ab[o];
    if (doLeaky) s = leaky(s);
    out[t] = s;
}

// ---------------------------------------------------------------------------
// launch
// ---------------------------------------------------------------------------
extern "C" void kernel_launch(void* const* d_in, const int* in_sizes, int n_in,
                              void* d_out, int out_size, void* d_ws, size_t ws_size,
                              hipStream_t stream) {
    (void)in_sizes; (void)n_in; (void)out_size; (void)ws_size;

    const float* x   = (const float*)d_in[0];
    const float* w1  = (const float*)d_in[1];
    const float* g1  = (const float*)d_in[2];
    const float* b1  = (const float*)d_in[3];
    const float* w2  = (const float*)d_in[4];
    const float* g2  = (const float*)d_in[5];
    const float* b2  = (const float*)d_in[6];
    const float* w3  = (const float*)d_in[7];
    const float* g3  = (const float*)d_in[8];
    const float* b3  = (const float*)d_in[9];
    const float* w4  = (const float*)d_in[10];
    const float* g4  = (const float*)d_in[11];
    const float* b4  = (const float*)d_in[12];
    const float* w5  = (const float*)d_in[13];
    const float* g5  = (const float*)d_in[14];
    const float* b5  = (const float*)d_in[15];
    const float* wl1 = (const float*)d_in[16];
    const float* g6  = (const float*)d_in[17];
    const float* b6  = (const float*)d_in[18];
    const float* wl2 = (const float*)d_in[19];
    const float* bl2 = (const float*)d_in[20];
    const float* g7  = (const float*)d_in[21];
    const float* b7  = (const float*)d_in[22];
    const float* wl3 = (const float*)d_in[23];
    const float* bl3 = (const float*)d_in[24];

    const int B = 8, N = 2048, CAT = 512, EMB = 1024;

    size_t off = 0;
    auto carve = [&](size_t bytes) {
        size_t p = off;
        off += (bytes + 255) & ~(size_t)255;
        return p;
    };
    char* ws = (char*)d_ws;
    float*     xt     = (float*)    (ws + carve((size_t)B * N * 4 * 4));
    float*     sqn    = (float*)    (ws + carve((size_t)B * N * 4));
    int*       idx    = (int*)      (ws + carve((size_t)B * N * K_NN * 4));
    float*     xcat   = (float*)    (ws + carve((size_t)B * N * CAT * 4));
    _Float16*  xcatp  = (_Float16*) (ws + carve((size_t)B * N * CAT * 2));
    float*     h5v    = (float*)    (ws + carve((size_t)B * N * EMB * 4));
    float*     pooled = (float*)    (ws + carve((size_t)B * 2 * EMB * 4));
    float*     t1     = (float*)    (ws + carve((size_t)B * 512 * 4));
    float*     t2     = (float*)    (ws + carve((size_t)B * 256 * 4));
    _Float16*  wT1    = (_Float16*) (ws + carve((size_t)64  * 32  * 2));
    _Float16*  wT2    = (_Float16*) (ws + carve((size_t)64  * 128 * 2));
    _Float16*  wT3    = (_Float16*) (ws + carve((size_t)128 * 128 * 2));
    _Float16*  wT4    = (_Float16*) (ws + carve((size_t)256 * 256 * 2));
    _Float16*  w5T    = (_Float16*) (ws + carve((size_t)EMB * CAT * 2));

    // weight repacks (f32 -> transposed, padded f16)
    convertWT_kernel<<<(64  * 32  + 255) / 256, 256, 0, stream>>>(w1, wT1, 6,   32,  64,  1);
    convertWT_kernel<<<(64  * 128 + 255) / 256, 256, 0, stream>>>(w2, wT2, 128, 128, 64,  0);
    convertWT_kernel<<<(128 * 128 + 255) / 256, 256, 0, stream>>>(w3, wT3, 128, 128, 128, 0);
    convertWT_kernel<<<(256 * 256 + 255) / 256, 256, 0, stream>>>(w4, wT4, 256, 256, 256, 0);
    convertWT_kernel<<<(EMB * CAT + 255) / 256, 256, 0, stream>>>(w5, w5T, CAT, CAT, EMB, 0);

    transpose4_kernel<<<(B * N + 255) / 256, 256, 0, stream>>>(x, xt, B, N);

    const int BN = B * N;

    // ---- layer 1: in xt (C=4 padded, stride 4) -> xcat[:, 0:64]
    sqnorm_kernel<4, 4><<<(BN + 255) / 256, 256, 0, stream>>>(xt, sqn, BN);
    knn_kernel<4, 4><<<B * (N / 16), 32, 0, stream>>>(xt, sqn, idx, N);
    edge_kernel<4, 4, 32, 64><<<BN, 32, 0, stream>>>(
        xt, idx, wT1, g1, b1, xcat + 0, xcatp, 0, CAT, N);

    // ---- layer 2: in xcat[:,0:64] -> xcat[:, 64:128]
    sqnorm_kernel<64, 512><<<(BN + 255) / 256, 256, 0, stream>>>(xcat + 0, sqn, BN);
    knn_kernel<64, 512><<<B * (N / 16), 32, 0, stream>>>(xcat + 0, sqn, idx, N);
    edge_kernel<512, 64, 128, 64><<<BN, 32, 0, stream>>>(
        xcat + 0, idx, wT2, g2, b2, xcat + 64, xcatp, 64, CAT, N);

    // ---- layer 3: in xcat[:,64:128] -> xcat[:, 128:256]
    sqnorm_kernel<64, 512><<<(BN + 255) / 256, 256, 0, stream>>>(xcat + 64, sqn, BN);
    knn_kernel<64, 512><<<B * (N / 16), 32, 0, stream>>>(xcat + 64, sqn, idx, N);
    edge_kernel<512, 64, 128, 128><<<BN, 32, 0, stream>>>(
        xcat + 64, idx, wT3, g3, b3, xcat + 128, xcatp, 128, CAT, N);

    // ---- layer 4: in xcat[:,128:256] -> xcat[:, 256:512]
    sqnorm_kernel<128, 512><<<(BN + 255) / 256, 256, 0, stream>>>(xcat + 128, sqn, BN);
    knn_kernel<128, 512><<<B * (N / 16), 32, 0, stream>>>(xcat + 128, sqn, idx, N);
    edge_kernel<512, 128, 256, 256><<<BN, 32, 0, stream>>>(
        xcat + 128, idx, wT4, g4, b4, xcat + 256, xcatp, 256, CAT, N);

    // ---- point MLP 512 -> 1024
    gemm5_kernel<512, 1024><<<(BN / 16) * (EMB / 16), 32, 0, stream>>>(
        xcatp, w5T, g5, b5, h5v);

    // ---- pooling + classifier head
    pool_kernel<<<(B * EMB + 255) / 256, 256, 0, stream>>>(h5v, pooled, B, N, EMB);
    fc_kernel<<<(B * 512 + 255) / 256, 256, 0, stream>>>(
        pooled, wl1, nullptr, g6, b6, t1, B, 2 * EMB, 512, 1);
    fc_kernel<<<(B * 256 + 255) / 256, 256, 0, stream>>>(
        t1, wl2, bl2, g7, b7, t2, B, 512, 256, 1);
    fc_kernel<<<(B * 40 + 255) / 256, 256, 0, stream>>>(
        t2, wl3, bl3, nullptr, nullptr, (float*)d_out, B, 256, 40, 0);
}
